// MyModel_40819369181549
// MI455X (gfx1250) — matrix operand order repaired and checked by reading, not compiled, not run on hardware
//
#include <hip/hip_runtime.h>
#include <hip/hip_bf16.h>
#include <math.h>

typedef __attribute__((ext_vector_type(16))) _Float16 v16h;
typedef __attribute__((ext_vector_type(8)))  float    v8f;

#define B_  64
#define T_  1024
#define H_  64
#define NH_ 2
#define GD_ 192   /* 3*H */

// ---------------------------------------------------------------------------
// WMMA fragment loaders (gfx1250 wave32 layouts, CDNA5 ISA §7.12.2)
// ---------------------------------------------------------------------------

// A fragment: 16x32 f16 tile, rows row0..row0+15, K-chunk [k0, k0+32).
// lanes 0-15: M=lane, halves 0-15 -> K=k0+h
// lanes 16-31: M=lane-16, K=k0+16+h.
static __device__ __forceinline__ v16h load_a_frag(const _Float16* A, int row0, int lda, int k0) {
  int lane = threadIdx.x & 31;
  int m = lane & 15;
  int kb = (lane < 16) ? 0 : 16;
  const _Float16* p = A + (size_t)(row0 + m) * lda + k0;
  v16h a;
#pragma unroll
  for (int h = 0; h < 16; ++h) a[h] = p[kb + h];
  return a;
}

// B fragment for C = A @ W^T : B[k][n] = W[n0+n][k0+k], W row-major stride ldw.
// lane gives N = n0 + (lane&15); halves cover K = k0 + (lane<16?0:16) + h
// (contiguous in k -> vectorizes to b128 global loads).
static __device__ __forceinline__ v16h load_b_fragT(const _Float16* W, int n0, int ldw, int k0) {
  int lane = threadIdx.x & 31;
  int n = lane & 15;
  int kb = (lane < 16) ? 0 : 16;
  const _Float16* p = W + (size_t)(n0 + n) * ldw + k0 + kb;
  v16h b;
#pragma unroll
  for (int h = 0; h < 16; ++h) b[h] = p[h];
  return b;
}

static __device__ __forceinline__ v8f wmma_f16(v16h a, v16h b, v8f c) {
  return __builtin_amdgcn_wmma_f32_16x16x32_f16(false, a, false, b, (short)0, c, false, false);
}

// ---------------------------------------------------------------------------
// Weight conversion f32 -> f16 with column padding (so K is a multiple of 32)
// ---------------------------------------------------------------------------
__global__ void cvt_pad_f16(const float* __restrict__ src, _Float16* __restrict__ dst,
                            int rows, int cols, int dcols) {
  int i = blockIdx.x * blockDim.x + threadIdx.x;
  if (i >= rows * dcols) return;
  int r = i / dcols, c = i - r * dcols;
  dst[i] = (c < cols) ? (_Float16)src[r * cols + c] : (_Float16)0.f;
}

// ---------------------------------------------------------------------------
// Feature build: e_i | e_j | rbf(dist)  -> x16 (B*T, 32) f16 (cols 30,31 = 0)
// ---------------------------------------------------------------------------
__global__ void build_x(const int* __restrict__ atom_idx, const float* __restrict__ dist,
                        const float* __restrict__ emb, _Float16* __restrict__ x16) {
  int i = blockIdx.x * blockDim.x + threadIdx.x; // (b*T + t)
  if (i >= B_ * T_) return;
  int ii = atom_idx[i * 2 + 0];
  int jj = atom_idx[i * 2 + 1];
  float d = dist[i];
  bool mi = (ii != 0), mj = (jj != 0);
  _Float16* xp = x16 + (size_t)i * 32;
#pragma unroll
  for (int e = 0; e < 10; ++e) {
    float ei = mi ? emb[ii * 10 + e] : 0.f;
    float ej = mj ? emb[jj * 10 + e] : 0.f;
    float cc = (float)(e + 1) - d;            // GAMMA = 1
    float rb = mi ? __expf(-cc * cc) : 0.f;
    xp[e]      = (_Float16)ei;
    xp[10 + e] = (_Float16)ej;
    xp[20 + e] = (_Float16)rb;
  }
  xp[30] = (_Float16)0.f;
  xp[31] = (_Float16)0.f;
}

// ---------------------------------------------------------------------------
// Generic WMMA GEMM: C(MxN) = A(MxK) @ W^T + bias.  One wave per 16x16 tile.
// mode 0: f32 row-major out | mode 1: f16 row-major out
// mode 2: QKV scatter -- Q,K as (B*NH, T, 32); V TRANSPOSED as (B*NH, 32, T)
//         so the attention P@V B-fragment is contiguous in K.
// ---------------------------------------------------------------------------
__global__ __launch_bounds__(256) void gemm_wmma(
    const _Float16* __restrict__ A, const _Float16* __restrict__ W,
    const float* __restrict__ bias, int M, int N, int K, int mode,
    float* outf, _Float16* outh, _Float16* qb, _Float16* kb, _Float16* vb) {
  int gw = blockIdx.x * 8 + (threadIdx.x >> 5);
  int lane = threadIdx.x & 31;
  int ntn = N >> 4;
  int mt = gw / ntn, nt = gw - mt * ntn;
  if (mt * 16 >= M) return;                    // wave-uniform; EXEC stays full
  v8f c = {};
  for (int k0 = 0; k0 < K; k0 += 32) {
    v16h a = load_a_frag(A, mt * 16, K, k0);
    v16h b = load_b_fragT(W, nt * 16, K, k0);
    c = wmma_f16(a, b, c);
  }
  int n = nt * 16 + (lane & 15);
  float bv = bias[n];
  int mo = (lane < 16) ? 0 : 8;
#pragma unroll
  for (int r = 0; r < 8; ++r) {
    int m = mt * 16 + r + mo;
    float val = c[r] + bv;
    if (mode == 0) {
      outf[(size_t)m * N + n] = val;
    } else if (mode == 1) {
      outh[(size_t)m * N + n] = (_Float16)val;
    } else {
      int which = n >> 6, cc = n & 63, head = cc >> 5, dd = cc & 31;
      int b_ = m / T_, t = m - b_ * T_;
      size_t bh = (size_t)(b_ * NH_ + head);
      if (which == 0)      qb[(bh * T_ + t) * 32 + dd] = (_Float16)val;
      else if (which == 1) kb[(bh * T_ + t) * 32 + dd] = (_Float16)val;
      else                 vb[(bh * 32 + dd) * T_ + t] = (_Float16)val;  // V^T
    }
  }
}

// ---------------------------------------------------------------------------
// GRU layer: persistent recurrent loop. 4 blocks x 16 batch rows, 12 waves.
// Wave j owns N-tile j of gh = h @ w_hh^T (16x192); w_hh fragments live in
// registers for all 1024 steps; h ping-pongs through LDS (f16 for WMMA A,
// f32 for the exact state update).
// ---------------------------------------------------------------------------
__global__ __launch_bounds__(384) void gru_wmma(
    const float* __restrict__ xw, const _Float16* __restrict__ whh,
    const float* __restrict__ bhh, _Float16* __restrict__ gout) {
  __shared__ _Float16 h16[16 * 64];
  __shared__ float    hf[16 * 64];
  __shared__ float    gh[16 * GD_];
  int tid = threadIdx.x;
  int wave = tid >> 5, lane = tid & 31;
  int bbase = blockIdx.x * 16;
  for (int i = tid; i < 16 * 64; i += 384) { hf[i] = 0.f; h16[i] = (_Float16)0.f; }
  v16h wb0 = load_b_fragT(whh, wave * 16, 64, 0);    // persistent across t
  v16h wb1 = load_b_fragT(whh, wave * 16, 64, 32);
  __syncthreads();
  int n = lane & 15;
  int mo = (lane < 16) ? 0 : 8;
  for (int t = 0; t < T_; ++t) {
    v16h a0 = load_a_frag(h16, 0, 64, 0);
    v16h a1 = load_a_frag(h16, 0, 64, 32);
    v8f c = {};
    c = wmma_f16(a0, wb0, c);
    c = wmma_f16(a1, wb1, c);
#pragma unroll
    for (int r = 0; r < 8; ++r) gh[(r + mo) * GD_ + wave * 16 + n] = c[r];
    __syncthreads();
    for (int i = tid; i < 16 * 64; i += 384) {
      int row = i >> 6, u = i & 63;
      const float* xp = xw + (((size_t)(bbase + row)) * T_ + t) * GD_;
      if (t + 1 < T_) __builtin_prefetch(xp + GD_, 0, 1);   // next timestep
      float xr = xp[u], xz = xp[64 + u], xn = xp[128 + u];
      float hr = gh[row * GD_ + u]        + bhh[u];
      float hz = gh[row * GD_ + 64 + u]   + bhh[64 + u];
      float hn = gh[row * GD_ + 128 + u]  + bhh[128 + u];
      float rg = 1.f / (1.f + __expf(-(xr + hr)));
      float zg = 1.f / (1.f + __expf(-(xz + hz)));
      float ng = tanhf(xn + rg * hn);
      float hnew = (1.f - zg) * ng + zg * hf[i];
      hf[i] = hnew;
      h16[i] = (_Float16)hnew;
      gout[(((size_t)(bbase + row)) * T_ + t) * 64 + u] = (_Float16)hnew;
    }
    __syncthreads();
  }
}

// ---------------------------------------------------------------------------
// Flash attention: one wave per 16-query tile (head dim 32). Online softmax;
// D-layout row stats reduced with shfl_xor inside 16-lane halves; P goes
// through LDS to re-shape into an A fragment for P @ V.  V is pre-transposed
// (32 x T), so both K and V B-fragments are K-contiguous b128 loads.
// ---------------------------------------------------------------------------
__global__ __launch_bounds__(128) void attn_flash(
    const _Float16* __restrict__ q, const _Float16* __restrict__ k,
    const _Float16* __restrict__ vt, _Float16* __restrict__ out, float scale) {
  __shared__ _Float16 P[4][16 * 32];
  int wave = threadIdx.x >> 5, lane = threadIdx.x & 31;
  int gt = blockIdx.x * 4 + wave;
  int bh = gt >> 6;                 // T/16 = 64 q-tiles per (b,h)
  int q0 = (gt & 63) * 16;
  const _Float16* qp = q  + (size_t)bh * T_ * 32;
  const _Float16* kp = k  + (size_t)bh * T_ * 32;
  const _Float16* vp = vt + (size_t)bh * 32 * T_;   // (32, T) row-major
  v16h qa = load_a_frag(qp, q0, 32, 0);      // Q tile, persistent
  v8f o0 = {}, o1 = {};
  float mst[8], lst[8];
#pragma unroll
  for (int r = 0; r < 8; ++r) { mst[r] = -1e30f; lst[r] = 0.f; }
  _Float16* Pl = &P[wave][0];
  int n = lane & 15, mo = (lane < 16) ? 0 : 8;
  for (int k0 = 0; k0 < T_; k0 += 32) {
    v16h kb0 = load_b_fragT(kp, k0, 32, 0);       // B[d][n] = K[k0+n][d]
    v16h kb1 = load_b_fragT(kp, k0 + 16, 32, 0);
    v8f s0 = {}, s1 = {};
    s0 = wmma_f16(qa, kb0, s0);
    s1 = wmma_f16(qa, kb1, s1);
#pragma unroll
    for (int r = 0; r < 8; ++r) {
      float a = s0[r] * scale, b = s1[r] * scale;
      float mx = fmaxf(a, b);
      mx = fmaxf(mx, __shfl_xor(mx, 1, 32));
      mx = fmaxf(mx, __shfl_xor(mx, 2, 32));
      mx = fmaxf(mx, __shfl_xor(mx, 4, 32));
      mx = fmaxf(mx, __shfl_xor(mx, 8, 32));
      float mnew  = fmaxf(mst[r], mx);
      float alpha = __expf(mst[r] - mnew);
      float p0 = __expf(a - mnew), p1 = __expf(b - mnew);
      float rs = p0 + p1;
      rs += __shfl_xor(rs, 1, 32);
      rs += __shfl_xor(rs, 2, 32);
      rs += __shfl_xor(rs, 4, 32);
      rs += __shfl_xor(rs, 8, 32);
      lst[r] = alpha * lst[r] + rs;
      mst[r] = mnew;
      o0[r] *= alpha; o1[r] *= alpha;
      Pl[(r + mo) * 32 + n]      = (_Float16)p0;   // D-layout -> (row,col) LDS
      Pl[(r + mo) * 32 + 16 + n] = (_Float16)p1;
    }
    v16h pa  = load_a_frag(Pl, 0, 32, 0);          // same-wave DS in-order
    // P@V: B[k][n] = V[k0+k][n] = Vt[n][k0+k] -> K-contiguous fragment
    v16h vb0 = load_b_fragT(vp, 0,  T_, k0);
    v16h vb1 = load_b_fragT(vp, 16, T_, k0);
    o0 = wmma_f16(pa, vb0, o0);
    o1 = wmma_f16(pa, vb1, o1);
  }
  int b_ = bh >> 1, head = bh & 1;
#pragma unroll
  for (int r = 0; r < 8; ++r) {
    float inv = 1.f / lst[r];
    size_t row = (size_t)b_ * T_ + q0 + r + mo;
    out[row * 64 + head * 32 + n]      = (_Float16)(o0[r] * inv);
    out[row * 64 + head * 32 + 16 + n] = (_Float16)(o1[r] * inv);
  }
}

// ---------------------------------------------------------------------------
// Final MLP on last timestep: relu(last @ w_l^T + b_l) @ w_l1^T + b_l1
// ---------------------------------------------------------------------------
__global__ void final_mlp(const _Float16* __restrict__ a2, const float* __restrict__ wl,
                          const float* __restrict__ bl, const float* __restrict__ wl1,
                          const float* __restrict__ bl1, float* __restrict__ out) {
  int b = blockIdx.x * blockDim.x + threadIdx.x;
  if (b >= B_) return;
  float last[64];
  const _Float16* p = a2 + ((size_t)b * T_ + (T_ - 1)) * 64;
#pragma unroll
  for (int u = 0; u < 64; ++u) last[u] = (float)p[u];
  float acc = bl1[0];
  for (int j = 0; j < 32; ++j) {
    float hj = bl[j];
    for (int u = 0; u < 64; ++u) hj += wl[j * 64 + u] * last[u];
    acc += wl1[j] * fmaxf(hj, 0.f);
  }
  out[b] = acc;
}

// ---------------------------------------------------------------------------
extern "C" void kernel_launch(void* const* d_in, const int* in_sizes, int n_in,
                              void* d_out, int out_size, void* d_ws, size_t ws_size,
                              hipStream_t stream) {
  const int*   atom_idx = (const int*)  d_in[0];
  const float* bond     = (const float*)d_in[1];
  const float* emb      = (const float*)d_in[2];
  const float* w_ih0    = (const float*)d_in[3];
  const float* w_hh0    = (const float*)d_in[4];
  const float* b_ih0    = (const float*)d_in[5];
  const float* b_hh0    = (const float*)d_in[6];
  const float* w_ih1    = (const float*)d_in[7];
  const float* w_hh1    = (const float*)d_in[8];
  const float* b_ih1    = (const float*)d_in[9];
  const float* b_hh1    = (const float*)d_in[10];
  const float* in_w1    = (const float*)d_in[11];
  const float* in_b1    = (const float*)d_in[12];
  const float* out_w1   = (const float*)d_in[13];
  const float* out_b1   = (const float*)d_in[14];
  const float* in_w2    = (const float*)d_in[15];
  const float* in_b2    = (const float*)d_in[16];
  const float* out_w2   = (const float*)d_in[17];
  const float* out_b2   = (const float*)d_in[18];
  const float* w_l      = (const float*)d_in[19];
  const float* b_l      = (const float*)d_in[20];
  const float* w_l1     = (const float*)d_in[21];
  const float* b_l1     = (const float*)d_in[22];

  char* ws = (char*)d_ws;
  size_t off = 0;
  auto alloc = [&](size_t bytes) {
    void* p = ws + off;
    off = (off + bytes + 255) & ~(size_t)255;
    return p;
  };
  _Float16* p_wih0 = (_Float16*)alloc((size_t)192 * 32 * 2);
  _Float16* p_whh0 = (_Float16*)alloc((size_t)192 * 64 * 2);
  _Float16* p_wih1 = (_Float16*)alloc((size_t)192 * 64 * 2);
  _Float16* p_whh1 = (_Float16*)alloc((size_t)192 * 64 * 2);
  _Float16* p_inw1 = (_Float16*)alloc((size_t)192 * 64 * 2);
  _Float16* p_ow1  = (_Float16*)alloc((size_t)64  * 64 * 2);
  _Float16* p_inw2 = (_Float16*)alloc((size_t)192 * 64 * 2);
  _Float16* p_ow2  = (_Float16*)alloc((size_t)64  * 64 * 2);
  _Float16* x16    = (_Float16*)alloc((size_t)B_ * T_ * 32 * 2);
  float*    xw     = (float*)   alloc((size_t)B_ * T_ * GD_ * 4);
  _Float16* gbuf   = (_Float16*)alloc((size_t)B_ * T_ * 64 * 2);
  _Float16* abuf   = (_Float16*)alloc((size_t)B_ * T_ * 64 * 2);
  _Float16* qb     = (_Float16*)alloc((size_t)B_ * NH_ * T_ * 32 * 2);
  _Float16* kb     = (_Float16*)alloc((size_t)B_ * NH_ * T_ * 32 * 2);
  _Float16* vb     = (_Float16*)alloc((size_t)B_ * NH_ * T_ * 32 * 2);  // V^T

  auto cvt = [&](const float* s, _Float16* d, int r, int c, int dc) {
    int tot = r * dc;
    cvt_pad_f16<<<(tot + 255) / 256, 256, 0, stream>>>(s, d, r, c, dc);
  };
  cvt(w_ih0, p_wih0, 192, 30, 32);
  cvt(w_hh0, p_whh0, 192, 64, 64);
  cvt(w_ih1, p_wih1, 192, 64, 64);
  cvt(w_hh1, p_whh1, 192, 64, 64);
  cvt(in_w1, p_inw1, 192, 64, 64);
  cvt(out_w1, p_ow1,  64, 64, 64);
  cvt(in_w2, p_inw2, 192, 64, 64);
  cvt(out_w2, p_ow2,  64, 64, 64);

  build_x<<<(B_ * T_ + 255) / 256, 256, 0, stream>>>(atom_idx, bond, emb, x16);

  const int M = B_ * T_;
  auto gemm = [&](const _Float16* A, const _Float16* W, const float* bias,
                  int N, int K, int mode, float* of, _Float16* oh) {
    int tiles = (M / 16) * (N / 16);
    gemm_wmma<<<(tiles + 7) / 8, 256, 0, stream>>>(A, W, bias, M, N, K, mode,
                                                   of, oh, qb, kb, vb);
  };

  // layer 0 GRU
  gemm(x16,  p_wih0, b_ih0, GD_, 32, 0, xw, nullptr);
  gru_wmma<<<4, 384, 0, stream>>>(xw, p_whh0, b_hh0, gbuf);
  // layer 1 GRU
  gemm(gbuf, p_wih1, b_ih1, GD_, 64, 0, xw, nullptr);
  gru_wmma<<<4, 384, 0, stream>>>(xw, p_whh1, b_hh1, gbuf);

  const float scale = 0.17677669529663687f;   // 1/sqrt(32)
  // MHA 1
  gemm(gbuf, p_inw1, in_b1, GD_, 64, 2, nullptr, nullptr);
  attn_flash<<<(B_ * NH_ * (T_ / 16)) / 4, 128, 0, stream>>>(qb, kb, vb, abuf, scale);
  gemm(abuf, p_ow1, out_b1, 64, 64, 1, nullptr, gbuf);
  // MHA 2
  gemm(gbuf, p_inw2, in_b2, GD_, 64, 2, nullptr, nullptr);
  attn_flash<<<(B_ * NH_ * (T_ / 16)) / 4, 128, 0, stream>>>(qb, kb, vb, abuf, scale);
  gemm(abuf, p_ow2, out_b2, 64, 64, 1, nullptr, gbuf);

  final_mlp<<<1, 64, 0, stream>>>(gbuf, w_l, b_l, w_l1, b_l1, (float*)d_out);
}
